// MultiLoRALayerMaskingHomEfficient_54623394070736
// MI455X (gfx1250) — matrix operand order repaired and checked by reading, not compiled, not run on hardware
//
#include <hip/hip_runtime.h>
#include <hip/hip_bf16.h>

// ---- CDNA5 WMMA vector types -------------------------------------------------
typedef __attribute__((ext_vector_type(16))) __bf16 v16bf;
typedef __attribute__((ext_vector_type(8)))  __bf16 v8bf;
typedef __attribute__((ext_vector_type(4)))  __bf16 v4bf;
typedef __attribute__((ext_vector_type(8)))  float  v8f;

#define N_BATCH   16
#define N_TOK     2048
#define DIM       4096      // in == out features
#define RANK      64
#define KC        256       // k-chunk staged in LDS for stage 1
#define SB_STRIDE (KC + 8)  // padded bf16 row stride (banks shift 4/row)
#define SBX_STRIDE (RANK + 8)
#define SA2_STRIDE (RANK + 8)
#define ROWS_PER_BLOCK 128  // 8 waves x 16 rows
#define SCALING   (1.0f / 64.0f)

__device__ __forceinline__ __bf16 bf(float f) { return (__bf16)f; }

__global__ __launch_bounds__(256)
void lora_fused_bf16_wmma(const float* __restrict__ x,        // [16][2048][4096]
                          const int*   __restrict__ adapter_ids, // [16]
                          const float* __restrict__ A,        // [64][4096][64]  (o, r)
                          const float* __restrict__ Bm,       // [64][64][4096]  (r, i)
                          float* __restrict__ out) {          // [16][2048][4096]
    __shared__ __align__(16) __bf16 sB [RANK * SB_STRIDE];          // 33,792 B (aliased by sA2)
    __shared__ __align__(16) __bf16 sBx[ROWS_PER_BLOCK * SBX_STRIDE]; // 18,432 B
    __bf16* sA2 = sB;                                               // stage-2 reuses stage-1 buffer

    const int tid  = threadIdx.x;
    const int wave = tid >> 5;
    const int lane = tid & 31;
    const int lo   = lane & 15;     // N index / M index within fragments
    const int hi   = lane >> 4;     // half-wave select
    const int sel  = hi * 8;        // A-frag K sub-offset

    const int b    = blockIdx.y;
    const int row0 = blockIdx.x * ROWS_PER_BLOCK;
    const int aid  = adapter_ids[b];

    const float* xb   = x   + (size_t)b   * N_TOK * DIM;
    const float* Bq   = Bm  + (size_t)aid * RANK  * DIM;   // row r contiguous in i
    const float* Aq   = A   + (size_t)aid * DIM   * RANK;  // row o contiguous in r
    float*       outb = out + (size_t)b   * N_TOK * DIM;

    // ============ Stage 1: Bx[rows, r] = x[rows, :] * B^T  (K = 4096) ============
    v8f acc[4] = {v8f{}, v8f{}, v8f{}, v8f{}};   // 4 rank-tiles of 16
    const int    myRow = row0 + wave * 16 + lo;  // this lane's A-matrix row
    const float* xrow  = xb + (size_t)myRow * DIM;

    for (int kc = 0; kc < DIM; kc += KC) {
        // Cooperative: B_req[0:64][kc:kc+KC] f32 -> bf16 LDS (once per workgroup)
        for (int i = tid; i < RANK * (KC / 4); i += 256) {
            const int r  = i >> 6;           // KC/4 == 64 quads per row
            const int kq = (i & 63) * 4;
            const float4 f = *(const float4*)(Bq + (size_t)r * DIM + kc + kq);
            v4bf v; v[0] = bf(f.x); v[1] = bf(f.y); v[2] = bf(f.z); v[3] = bf(f.w);
            *(v4bf*)(&sB[r * SB_STRIDE + kq]) = v;
        }
        __syncthreads();

        for (int ks = 0; ks < KC; ks += 32) {
            // A-fragment from global x, converted in registers.
            // Layout: lanes 0-15 hold K {0..7,16..23}, lanes 16-31 hold K {8..15,24..31}
            const float* p = xrow + kc + ks;
            const float4 fa = *(const float4*)(p + sel);
            const float4 fb = *(const float4*)(p + sel + 4);
            const float4 fc = *(const float4*)(p + sel + 16);
            const float4 fd = *(const float4*)(p + sel + 20);
            v16bf afrag;
            afrag[0]  = bf(fa.x); afrag[1]  = bf(fa.y); afrag[2]  = bf(fa.z); afrag[3]  = bf(fa.w);
            afrag[4]  = bf(fb.x); afrag[5]  = bf(fb.y); afrag[6]  = bf(fb.z); afrag[7]  = bf(fb.w);
            afrag[8]  = bf(fc.x); afrag[9]  = bf(fc.y); afrag[10] = bf(fc.z); afrag[11] = bf(fc.w);
            afrag[12] = bf(fd.x); afrag[13] = bf(fd.y); afrag[14] = bf(fd.z); afrag[15] = bf(fd.w);

            // B-fragment: column N=r of B^T == row r of B_req (contiguous in LDS).
            // Lanes 0-15: K {0..7,16..23} ; lanes 16-31: K {8..15,24..31}
            const int kk = ks + sel;
            #pragma unroll
            for (int rt = 0; rt < 4; ++rt) {
                const v8bf bl = *(const v8bf*)(&sB[(rt * 16 + lo) * SB_STRIDE + kk]);
                const v8bf bu = *(const v8bf*)(&sB[(rt * 16 + lo) * SB_STRIDE + kk + 16]);
                const v16bf bfrag = __builtin_shufflevector(bl, bu, 0,1,2,3,4,5,6,7,8,9,10,11,12,13,14,15);
                acc[rt] = __builtin_amdgcn_wmma_f32_16x16x32_bf16(
                    false, afrag, false, bfrag, (short)0, acc[rt], false, false);
            }
        }
        __syncthreads();   // before next chunk overwrites sB
    }

    // Scale by 1/64, convert to bf16, park Bx in LDS to re-shape for stage 2.
    // C layout: c[j] -> (M = j + 8*hi, N = lo)
    #pragma unroll
    for (int rt = 0; rt < 4; ++rt) {
        #pragma unroll
        for (int j = 0; j < 8; ++j) {
            const int rr = wave * 16 + j + 8 * hi;   // local row 0..127
            sBx[rr * SBX_STRIDE + rt * 16 + lo] = bf(acc[rt][j] * SCALING);
        }
    }
    __syncthreads();

    // ============ Stage 2: out[rows, o] = Bx[rows, :] * A^T  (K = 64) ============
    // A-fragments (Bx) held in registers for the entire output sweep.
    const int arow = (wave * 16 + lo) * SBX_STRIDE;
    v16bf a2[2];
    #pragma unroll
    for (int h = 0; h < 2; ++h) {
        const v8bf l = *(const v8bf*)(&sBx[arow + h * 32 + sel]);
        const v8bf u = *(const v8bf*)(&sBx[arow + h * 32 + 16 + sel]);
        a2[h] = __builtin_shufflevector(l, u, 0,1,2,3,4,5,6,7,8,9,10,11,12,13,14,15);
    }

    for (int oc = 0; oc < DIM; oc += 64) {
        // Cooperative: A_req[oc:oc+64][0:64] f32 -> bf16 LDS (once per workgroup)
        for (int i = tid; i < 64 * (RANK / 4); i += 256) {
            const int o  = i >> 4;
            const int rq = (i & 15) * 4;
            const float4 f = *(const float4*)(Aq + (size_t)(oc + o) * RANK + rq);
            v4bf v; v[0] = bf(f.x); v[1] = bf(f.y); v[2] = bf(f.z); v[3] = bf(f.w);
            *(v4bf*)(&sA2[o * SA2_STRIDE + rq]) = v;
        }
        __syncthreads();

        #pragma unroll
        for (int ot = 0; ot < 4; ++ot) {
            v8f c = {};
            #pragma unroll
            for (int h = 0; h < 2; ++h) {
                // column N=o of A^T == row o of A_req (contiguous in LDS)
                const v8bf bl = *(const v8bf*)(&sA2[(ot * 16 + lo) * SA2_STRIDE + h * 32 + sel]);
                const v8bf bu = *(const v8bf*)(&sA2[(ot * 16 + lo) * SA2_STRIDE + h * 32 + 16 + sel]);
                const v16bf bfrag = __builtin_shufflevector(bl, bu, 0,1,2,3,4,5,6,7,8,9,10,11,12,13,14,15);
                c = __builtin_amdgcn_wmma_f32_16x16x32_bf16(
                    false, a2[h], false, bfrag, (short)0, c, false, false);
            }
            // Store 16x16 f32 tile: lanes 0-15 -> row j, lanes 16-31 -> row j+8
            float* po = outb + (size_t)(row0 + wave * 16 + 8 * hi) * DIM + (oc + ot * 16 + lo);
            #pragma unroll
            for (int j = 0; j < 8; ++j) {
                po[(size_t)j * DIM] = c[j];
            }
        }
        __syncthreads();   // before next chunk overwrites sA2
    }
}

extern "C" void kernel_launch(void* const* d_in, const int* in_sizes, int n_in,
                              void* d_out, int out_size, void* d_ws, size_t ws_size,
                              hipStream_t stream) {
    const float* x   = (const float*)d_in[0];
    const int*   ids = (const int*)d_in[1];
    const float* A   = (const float*)d_in[2];
    const float* Bm  = (const float*)d_in[3];
    float*       out = (float*)d_out;
    (void)in_sizes; (void)n_in; (void)d_ws; (void)ws_size; (void)out_size;

    dim3 grid(N_TOK / ROWS_PER_BLOCK, N_BATCH);   // 16 x 16
    dim3 block(256);                              // 8 wave32
    lora_fused_bf16_wmma<<<grid, block, 0, stream>>>(x, ids, A, Bm, out);
}
